// UltraControllableGNN_30631706755219
// MI455X (gfx1250) — matrix operand order repaired and checked by reading, not compiled, not run on hardware
//
#include <hip/hip_runtime.h>
#include <cmath>

// ---------------- problem constants ----------------
#define BATCH   8
#define NNODES  512
#define LTOK    128
#define D_MODEL 512
#define HEADS   8
#define HDIM    64
#define LLMD    1536
#define FFD     2048

typedef _Float16 v16h __attribute__((ext_vector_type(16)));
typedef _Float16 v8h  __attribute__((ext_vector_type(8)));
typedef float    v8f  __attribute__((ext_vector_type(8)));

// pointer payload type for the async b128 builtin: int4
typedef int i4 __attribute__((vector_size(16)));
typedef __attribute__((address_space(1))) i4 gi4;   // global int4
typedef __attribute__((address_space(3))) i4 li4;   // LDS int4

#define ACT_NONE 0
#define ACT_GELU 1
#define ACT_GATE 2

// CDNA5 async global->LDS copy (ASYNCcnt) with safe fallback.
#if defined(__AMDGCN__) && __has_builtin(__builtin_amdgcn_global_load_async_to_lds_b128)
#define USE_ASYNC_LDS 1
#else
#define USE_ASYNC_LDS 0
#endif

__device__ __forceinline__ void cp16_async(const _Float16* g, _Float16* l) {
#if USE_ASYNC_LDS
    __builtin_amdgcn_global_load_async_to_lds_b128((gi4*)g, (li4*)l, 0, 0);
#else
    *(v8h*)l = *(const v8h*)g;   // vectorized synchronous fallback
#endif
}

template <int CNT>
__device__ __forceinline__ void cp_wait_async() {
#if USE_ASYNC_LDS
#if __has_builtin(__builtin_amdgcn_s_wait_asynccnt)
    __builtin_amdgcn_s_wait_asynccnt(CNT);
#else
    asm volatile("s_wait_asynccnt %0" :: "i"(CNT) : "memory");
#endif
#endif
}

// =====================================================================
// f16-native batched WMMA GEMM:
//   A: f16 row-major [M][K] (lda), B: f16 [N][K] (ldb, K-innermost!)
//   out = act(alpha * A@B^T_layout + bias):
//     C   (f32, optional), Ch (f16 row-major mirror, optional),
//     ChT (f16 transposed mirror [batch][col][tok], ctTok>0, optional)
// Block = 256 threads = 8 wave32s; tile 128x64; K chunks of 32 staged to
// double-buffered LDS via GLOBAL_LOAD_ASYNC_TO_LDS_B128 so the DMA for
// chunk i+1 overlaps the WMMAs of chunk i (wait asynccnt<=3 = new chunk
// in flight, old chunk retired).
// z -> (zb=z/Hdim, zh=z%Hdim) with independent operand strides.
// =====================================================================
__global__ __launch_bounds__(256) void gemm_wmma_kernel(
    const _Float16* __restrict__ A, const _Float16* __restrict__ B,
    const float* __restrict__ bias, const float* __restrict__ aux,
    float* __restrict__ C, _Float16* __restrict__ Ch, _Float16* __restrict__ ChT,
    int K, int lda, int ldb, int ldc, int ctTok,
    long sAb, long sBb, long sCb, int Hdim, long sAh, long sBh, long sCh,
    float alpha, int act)
{
    __shared__ _Float16 As[2][128][32];   // 2 x 8 KB, [m][k]
    __shared__ _Float16 Bst[2][64][32];   // 2 x 4 KB, [n][k]

    const int tid  = threadIdx.x;
    const int lane = tid & 31;
    const int wave = tid >> 5;
    const int half = lane >> 4;
    const int l15  = lane & 15;

    const int n0 = blockIdx.x * 64;
    const int m0 = blockIdx.y * 128;
    const int z  = blockIdx.z;
    const int zb = z / Hdim, zh = z % Hdim;

    const _Float16* Ab = A + (size_t)zb * sAb + (size_t)zh * sAh;
    const _Float16* Bb = B + (size_t)zb * sBb + (size_t)zh * sBh;

    v8f acc[4] = {};

    // staging assignments (16B segments): A = 512 segs (2/thread), B = 256 segs
    const int ra0 = tid >> 2,          sa0 = tid & 3;          // A rows 0..63
    const int ra1 = 64 + (tid >> 2),   sa1 = tid & 3;          // A rows 64..127
    const int rb  = tid >> 2,          sb  = tid & 3;          // B rows 0..63

    auto issueChunk = [&](int buf, int k0) {
        cp16_async(Ab + (size_t)(m0 + ra0) * lda + k0 + sa0 * 8, &As[buf][ra0][sa0 * 8]);
        cp16_async(Ab + (size_t)(m0 + ra1) * lda + k0 + sa1 * 8, &As[buf][ra1][sa1 * 8]);
        cp16_async(Bb + (size_t)(n0 + rb) * ldb + k0 + sb * 8, &Bst[buf][rb][sb * 8]);
    };

    issueChunk(0, 0);
    int cur = 0;
    for (int k0 = 0; k0 < K; k0 += 32) {
        if (k0 + 32 < K) {
            issueChunk(cur ^ 1, k0 + 32);      // prior barrier: buf free
            if (k0 + 64 < K && tid < 128)      // L2 prefetch two chunks ahead
                __builtin_prefetch(Ab + (size_t)(m0 + tid) * lda + k0 + 64, 0, 0);
            cp_wait_async<3>();                // chunk k0 landed; k0+32 in flight
        } else {
            cp_wait_async<0>();
        }
        __syncthreads();

        // A fragment (16x32): lanes<16 K 0..7 & 16..23, lanes>=16 K 8..15 & 24..31
        union { v16h v; v8h h[2]; } af;
        af.h[0] = *(const v8h*)&As[cur][wave * 16 + l15][half * 8];
        af.h[1] = *(const v8h*)&As[cur][wave * 16 + l15][16 + half * 8];

        #pragma unroll
        for (int t = 0; t < 4; ++t) {   // B fragment (32x16): lanes<16 K 0..7 & 16..23, lanes>=16 K 8..15 & 24..31
            union { v16h v; v8h h[2]; } bf;
            bf.h[0] = *(const v8h*)&Bst[cur][t * 16 + l15][half * 8];
            bf.h[1] = *(const v8h*)&Bst[cur][t * 16 + l15][16 + half * 8];
            acc[t] = __builtin_amdgcn_wmma_f32_16x16x32_f16(
                false, af.v, false, bf.v, (short)0, acc[t], false, false);
        }
        __syncthreads();                       // all reads of buf[cur] done
        cur ^= 1;
    }

    float*     Cb  = C   ? C   + (size_t)zb * sCb + (size_t)zh * sCh : nullptr;
    _Float16*  Chb = Ch  ? Ch  + (size_t)zb * sCb + (size_t)zh * sCh : nullptr;
    const float* Xb = aux ? aux + (size_t)zb * sCb + (size_t)zh * sCh : nullptr;

    const int rowBase = m0 + wave * 16 + half * 8;
    #pragma unroll
    for (int t = 0; t < 4; ++t) {
        const int col = n0 + t * 16 + l15;
        const float bval = bias ? bias[col] : 0.0f;
        #pragma unroll
        for (int r = 0; r < 8; ++r) {
            const int row = rowBase + r;
            float v = acc[t][r] * alpha + bval;
            if (act == ACT_GELU) {
                v = 0.5f * v * (1.0f + erff(v * 0.70710678118654752f));
            } else if (act == ACT_GATE) {
                float gate = 1.0f / (1.0f + __expf(-v));
                v = (1.0f - gate) * Xb[(size_t)row * ldc + col];
            }
            if (Cb)  Cb[(size_t)row * ldc + col] = v;
            if (Chb) Chb[(size_t)row * ldc + col] = (_Float16)v;
            if (ChT) {      // pre-transposed mirror: [tok-batch][col][tok]
                int bId = row / ctTok, tId = row % ctTok;
                ChT[((size_t)bId * ldc + col) * ctTok + tId] = (_Float16)v;
            }
        }
    }
}

// ---------- f32 -> f16 (same layout) ----------
__global__ __launch_bounds__(256) void cvt_kernel(
    const float* __restrict__ src, _Float16* __restrict__ dst, long n)
{
    long i = (long)blockIdx.x * 256 + threadIdx.x;
    if (i < n) dst[i] = (_Float16)src[i];
}

// ---------- weight f32 [K][N] (ldw) -> f16 [N][K] ----------
__global__ __launch_bounds__(256) void cvtT_kernel(
    const float* __restrict__ W, _Float16* __restrict__ out, int K, int N, int ldw)
{
    long i = (long)blockIdx.x * 256 + threadIdx.x;
    if (i < (long)K * N) {
        int n = (int)(i / K), k = (int)(i % K);
        out[i] = (_Float16)W[(size_t)k * ldw + n];
    }
}

// ---------- maxd[b] = max cdist ----------
__global__ __launch_bounds__(256) void maxdist_kernel(
    const float* __restrict__ coords, float* __restrict__ maxd)
{
    __shared__ float cs[NNODES * 2];
    __shared__ float red[256];
    const int b = blockIdx.x, tid = threadIdx.x;
    for (int i = tid; i < NNODES * 2; i += 256)
        cs[i] = coords[(size_t)b * NNODES * 2 + i];
    __syncthreads();
    float m2 = 0.0f;
    for (int p = tid; p < NNODES * NNODES; p += 256) {
        int i = p >> 9, j = p & (NNODES - 1);
        float dx = cs[2 * i] - cs[2 * j];
        float dy = cs[2 * i + 1] - cs[2 * j + 1];
        m2 = fmaxf(m2, dx * dx + dy * dy);
    }
    red[tid] = m2; __syncthreads();
    for (int s = 128; s > 0; s >>= 1) {
        if (tid < s) red[tid] = fmaxf(red[tid], red[tid + s]);
        __syncthreads();
    }
    if (tid == 0) maxd[b] = sqrtf(fmaxf(red[0], 1e-12f));
}

// ---------- logits += dist_bias - pen (broadcast over heads) ----------
__global__ __launch_bounds__(128) void geom_bias_kernel(
    float* __restrict__ logits, const float* __restrict__ coords,
    const float* __restrict__ maxd, const float* __restrict__ cp_ptr)
{
    const int b = blockIdx.x / NNODES;
    const int i = blockIdx.x % NNODES;
    const float xi = coords[((size_t)b * NNODES + i) * 2];
    const float yi = coords[((size_t)b * NNODES + i) * 2 + 1];
    const float md = maxd[b];
    const float cp = cp_ptr[0];
    const float bm = (b < 4) ? 1.0f : 0.0f;
    for (int j = threadIdx.x; j < NNODES; j += 128) {
        float dx = xi - coords[((size_t)b * NNODES + j) * 2];
        float dy = yi - coords[((size_t)b * NNODES + j) * 2 + 1];
        float d  = sqrtf(fmaxf(dx * dx + dy * dy, 1e-12f));
        float add = (1.0f - d / (md + 1e-8f)) - ((d > 0.7f * md) ? cp * bm : 0.0f);
        #pragma unroll
        for (int h = 0; h < HEADS; ++h)
            logits[((size_t)(b * HEADS + h) * NNODES + i) * NNODES + j] += add;
    }
}

// ---------- row softmax (in place) + f16 mirror ----------
__global__ __launch_bounds__(128) void softmax_kernel(
    float* __restrict__ data, int R, _Float16* __restrict__ mirror)
{
    __shared__ float red[128];
    float* p = data + (size_t)blockIdx.x * R;
    _Float16* ph = mirror + (size_t)blockIdx.x * R;
    const int tid = threadIdx.x;
    float lmax = -1e30f;
    for (int j = tid; j < R; j += 128) lmax = fmaxf(lmax, p[j]);
    red[tid] = lmax; __syncthreads();
    for (int s = 64; s > 0; s >>= 1) {
        if (tid < s) red[tid] = fmaxf(red[tid], red[tid + s]);
        __syncthreads();
    }
    const float m = red[0]; __syncthreads();
    float lsum = 0.0f;
    for (int j = tid; j < R; j += 128) {
        float e = __expf(p[j] - m);
        p[j] = e; lsum += e;
    }
    red[tid] = lsum; __syncthreads();
    for (int s = 64; s > 0; s >>= 1) {
        if (tid < s) red[tid] += red[tid + s];
        __syncthreads();
    }
    const float inv = 1.0f / red[0];
    for (int j = tid; j < R; j += 128) {
        float v = p[j] * inv;
        p[j] = v;
        ph[j] = (_Float16)v;
    }
}

// ---------- y = LN(x1+x2)*g+b over D=512 (+ optional f16 mirror) ----------
__global__ __launch_bounds__(128) void ln_res_kernel(
    const float* __restrict__ x1, const float* __restrict__ x2,
    const float* __restrict__ g, const float* __restrict__ b,
    float* __restrict__ y, _Float16* __restrict__ yh)
{
    __shared__ float red[128];
    const size_t row = blockIdx.x;
    const float* p1 = x1 + row * D_MODEL;
    const float* p2 = x2 + row * D_MODEL;
    const int tid = threadIdx.x;
    float vals[4]; float s = 0.0f;
    #pragma unroll
    for (int e = 0; e < 4; ++e) {
        float v = p1[tid + 128 * e] + p2[tid + 128 * e];
        vals[e] = v; s += v;
    }
    red[tid] = s; __syncthreads();
    for (int st = 64; st > 0; st >>= 1) {
        if (tid < st) red[tid] += red[tid + st];
        __syncthreads();
    }
    const float mean = red[0] * (1.0f / (float)D_MODEL);
    __syncthreads();
    float vs = 0.0f;
    #pragma unroll
    for (int e = 0; e < 4; ++e) { float d = vals[e] - mean; vs += d * d; }
    red[tid] = vs; __syncthreads();
    for (int st = 64; st > 0; st >>= 1) {
        if (tid < st) red[tid] += red[tid + st];
        __syncthreads();
    }
    const float inv = rsqrtf(red[0] * (1.0f / (float)D_MODEL) + 1e-5f);
    #pragma unroll
    for (int e = 0; e < 4; ++e) {
        int col = tid + 128 * e;
        float v = (vals[e] - mean) * inv * g[col] + b[col];
        y[row * D_MODEL + col] = v;
        if (yh) yh[row * D_MODEL + col] = (_Float16)v;
    }
}

// ---------- host-side launch helper ----------
static inline void gemm(hipStream_t s, const _Float16* A, const _Float16* B,
                        const float* bias, const float* aux,
                        float* C, _Float16* Ch, _Float16* ChT,
                        int M, int N, int K, int lda, int ldb, int ldc, int ctTok,
                        long sAb, long sBb, long sCb, int Hdim,
                        long sAh, long sBh, long sCh, int nbatch,
                        float alpha, int act)
{
    dim3 g(N / 64, M / 128, nbatch), blk(256);
    gemm_wmma_kernel<<<g, blk, 0, s>>>(A, B, bias, aux, C, Ch, ChT,
                                       K, lda, ldb, ldc, ctTok,
                                       sAb, sBb, sCb, Hdim, sAh, sBh, sCh,
                                       alpha, act);
}

extern "C" void kernel_launch(void* const* d_in, const int* in_sizes, int n_in,
                              void* d_out, int out_size, void* d_ws, size_t ws_size,
                              hipStream_t stream) {
    (void)in_sizes; (void)n_in; (void)out_size; (void)ws_size;

    const float* x      = (const float*)d_in[0];
    const float* coords = (const float*)d_in[1];
    const float* llm    = (const float*)d_in[2];
    const float* Wq = (const float*)d_in[3];  const float* bq = (const float*)d_in[4];
    const float* Wk = (const float*)d_in[5];  const float* bk = (const float*)d_in[6];
    const float* Wv = (const float*)d_in[7];  const float* bv = (const float*)d_in[8];
    const float* Wo = (const float*)d_in[9];  const float* bo = (const float*)d_in[10];
    const float* cp = (const float*)d_in[11];
    const float* ln1g = (const float*)d_in[12]; const float* ln1b = (const float*)d_in[13];
    const float* g2cW = (const float*)d_in[14]; const float* g2cb = (const float*)d_in[15];
    const float* l2cW = (const float*)d_in[16]; const float* l2cb = (const float*)d_in[17];
    const float* cmInW = (const float*)d_in[18]; const float* cmInb = (const float*)d_in[19];
    const float* cmOutW = (const float*)d_in[20]; const float* cmOutb = (const float*)d_in[21];
    const float* goutW = (const float*)d_in[22]; const float* goutb = (const float*)d_in[23];
    const float* gng = (const float*)d_in[24]; const float* gnb = (const float*)d_in[25];
    // d_in[26..29] = GRU weights: dead code (hnew never reaches the output)
    const float* gateW = (const float*)d_in[30]; const float* gateb = (const float*)d_in[31];
    const float* ff1W = (const float*)d_in[32]; const float* ff1b = (const float*)d_in[33];
    const float* ff2W = (const float*)d_in[34]; const float* ff2b = (const float*)d_in[35];
    const float* ln2g = (const float*)d_in[36]; const float* ln2b = (const float*)d_in[37];
    float* out = (float*)d_out;

    // -------------- workspace carve --------------
    char* wp = (char*)d_ws;
    auto alloc = [&](size_t bytes) { char* r = wp; wp += (bytes + 255) & ~(size_t)255; return r; };
    const long SZ  = (long)BATCH * NNODES * D_MODEL;   // 2,097,152
    const long SZL = (long)BATCH * LTOK  * D_MODEL;    //   524,288
    const long SBIG = (long)BATCH * HEADS * NNODES * NNODES; // 16,777,216

    float* big  = (float*)alloc(SBIG * 4);       // logits f32
    float* bufP = (float*)alloc(SZ * 4);         // projection scratch f32
    float* h1f  = (float*)alloc(SZ * 4);
    float* h2f  = (float*)alloc(SZ * 4);
    float* uxf  = (float*)alloc(SZ * 4);
    float* maxd = (float*)alloc(BATCH * 4);

    _Float16* xh   = (_Float16*)alloc(SZ * 2);
    _Float16* llmh = (_Float16*)alloc((long)BATCH * LTOK * LLMD * 2);
    _Float16* slotA = (_Float16*)alloc(SZ * 2);  // qh -> oh -> o2h
    _Float16* slotB = (_Float16*)alloc(SZ * 2);  // kh -> gch -> geh
    _Float16* slotC = (_Float16*)alloc(SZ * 2);  // vT -> qph -> uxh
    _Float16* hSlot = (_Float16*)alloc(SZ * 2);  // h1h -> h2h
    _Float16* lch  = (_Float16*)alloc(SZL * 2);
    _Float16* kph  = (_Float16*)alloc(SZL * 2);
    _Float16* vpT  = (_Float16*)alloc(SZL * 2);
    _Float16* Ph   = (_Float16*)alloc(SBIG * 2); // P mirror -> FFN intermediate

    _Float16* wqT   = (_Float16*)alloc(512 * 512 * 2);
    _Float16* wkT   = (_Float16*)alloc(512 * 512 * 2);
    _Float16* wvT   = (_Float16*)alloc(512 * 512 * 2);
    _Float16* woT   = (_Float16*)alloc(512 * 512 * 2);
    _Float16* g2cT  = (_Float16*)alloc(512 * 512 * 2);
    _Float16* l2cT  = (_Float16*)alloc(1536 * 512 * 2);
    _Float16* cmTq  = (_Float16*)alloc(512 * 512 * 2);
    _Float16* cmTk  = (_Float16*)alloc(512 * 512 * 2);
    _Float16* cmTv  = (_Float16*)alloc(512 * 512 * 2);
    _Float16* cmOutT= (_Float16*)alloc(512 * 512 * 2);
    _Float16* goutT = (_Float16*)alloc(512 * 512 * 2);
    _Float16* gateT = (_Float16*)alloc(512 * 512 * 2);
    _Float16* ff1T  = (_Float16*)alloc((long)FFD * 512 * 2);
    _Float16* ff2T  = (_Float16*)alloc((long)FFD * 512 * 2);

    const int MN = BATCH * NNODES;   // 4096
    const int ML = BATCH * LTOK;     // 1024
    const long sQb = (long)NNODES * D_MODEL;   // [B,N,D] batch stride
    const long sLb = (long)LTOK * D_MODEL;     // [B,L,D] batch stride
    const long sSh = (long)NNODES * NNODES;    // [B,H,N,N] head stride
    const long sS2h = (long)NNODES * LTOK;     // [B,H,N,L] head stride

    // ---- one-time f16 conversion of GEMM operands ----
    auto cvt = [&](const float* s, _Float16* d, long n) {
        cvt_kernel<<<dim3((unsigned)((n + 255) / 256)), dim3(256), 0, stream>>>(s, d, n);
    };
    auto cvtT = [&](const float* W, _Float16* o, int K, int N, int ldw) {
        long tot = (long)K * N;
        cvtT_kernel<<<dim3((unsigned)((tot + 255) / 256)), dim3(256), 0, stream>>>(W, o, K, N, ldw);
    };
    cvt(x, xh, SZ);
    cvt(llm, llmh, (long)BATCH * LTOK * LLMD);
    cvtT(Wq, wqT, 512, 512, 512);      cvtT(Wk, wkT, 512, 512, 512);
    cvtT(Wv, wvT, 512, 512, 512);      cvtT(Wo, woT, 512, 512, 512);
    cvtT(g2cW, g2cT, 512, 512, 512);   cvtT(l2cW, l2cT, 1536, 512, 512);
    cvtT(cmInW + 0,    cmTq, 512, 512, 1536);
    cvtT(cmInW + 512,  cmTk, 512, 512, 1536);
    cvtT(cmInW + 1024, cmTv, 512, 512, 1536);
    cvtT(cmOutW, cmOutT, 512, 512, 512);
    cvtT(goutW, goutT, 512, 512, 512);
    cvtT(gateW, gateT, 512, 512, 512);          // first D rows of [2D,D]
    cvtT(ff1W, ff1T, 512, FFD, FFD);
    cvtT(ff2W, ff2T, FFD, 512, 512);

    // ---- 1. geometric self-attention ----
    _Float16 *qh = slotA, *kh = slotB, *vT = slotC;
    gemm(stream, xh, wqT, bq, nullptr, nullptr, qh, nullptr,
         MN, 512, 512, 512, 512, 512, 0, 0,0,0, 1, 0,0,0, 1, 1.0f, ACT_NONE);
    gemm(stream, xh, wkT, bk, nullptr, nullptr, kh, nullptr,
         MN, 512, 512, 512, 512, 512, 0, 0,0,0, 1, 0,0,0, 1, 1.0f, ACT_NONE);
    gemm(stream, xh, wvT, bv, nullptr, nullptr, nullptr, vT,
         MN, 512, 512, 512, 512, 512, NNODES, 0,0,0, 1, 0,0,0, 1, 1.0f, ACT_NONE);
    maxdist_kernel<<<dim3(BATCH), dim3(256), 0, stream>>>(coords, maxd);

    // S[b,h] = 0.125 * q @ k^T   (k mirror is already [n][k])
    gemm(stream, qh, kh, nullptr, nullptr, big, nullptr, nullptr,
         NNODES, NNODES, HDIM, 512, 512, NNODES, 0,
         sQb, sQb, (long)HEADS * sSh, HEADS, HDIM, HDIM, sSh,
         BATCH * HEADS, 0.125f, ACT_NONE);
    geom_bias_kernel<<<dim3(BATCH * NNODES), dim3(128), 0, stream>>>(big, coords, maxd, cp);
    softmax_kernel<<<dim3(BATCH * HEADS * NNODES), dim3(128), 0, stream>>>(big, NNODES, Ph);

    // o = P @ v  via pre-transposed vT[b][c][tok]
    _Float16* oh = slotA;   // q consumed
    gemm(stream, Ph, vT, nullptr, nullptr, nullptr, oh, nullptr,
         NNODES, HDIM, NNODES, NNODES, NNODES, 512, 0,
         (long)HEADS * sSh, (long)D_MODEL * NNODES, sQb, HEADS,
         sSh, (long)HDIM * NNODES, HDIM,
         BATCH * HEADS, 1.0f, ACT_NONE);
    gemm(stream, oh, woT, bo, nullptr, bufP, nullptr, nullptr,
         MN, 512, 512, 512, 512, 512, 0, 0,0,0, 1, 0,0,0, 1, 1.0f, ACT_NONE);
    ln_res_kernel<<<dim3(MN), dim3(128), 0, stream>>>(x, bufP, ln1g, ln1b, h1f, hSlot);

    // ---- 2. cross-modal attention ----
    _Float16* gch = slotB;  // k consumed
    gemm(stream, hSlot, g2cT, g2cb, nullptr, nullptr, gch, nullptr,
         MN, 512, 512, 512, 512, 512, 0, 0,0,0, 1, 0,0,0, 1, 1.0f, ACT_NONE);
    gemm(stream, llmh, l2cT, l2cb, nullptr, nullptr, lch, nullptr,
         ML, 512, LLMD, LLMD, LLMD, 512, 0, 0,0,0, 1, 0,0,0, 1, 1.0f, ACT_NONE);
    _Float16* qph = slotC;  // vT consumed
    gemm(stream, gch, cmTq, cmInb, nullptr, nullptr, qph, nullptr,
         MN, 512, 512, 512, 512, 512, 0, 0,0,0, 1, 0,0,0, 1, 1.0f, ACT_NONE);
    gemm(stream, lch, cmTk, cmInb + 512, nullptr, nullptr, kph, nullptr,
         ML, 512, 512, 512, 512, 512, 0, 0,0,0, 1, 0,0,0, 1, 1.0f, ACT_NONE);
    gemm(stream, lch, cmTv, cmInb + 1024, nullptr, nullptr, nullptr, vpT,
         ML, 512, 512, 512, 512, 512, LTOK, 0,0,0, 1, 0,0,0, 1, 1.0f, ACT_NONE);

    // S2[b,h] = 0.125 * qp @ kp^T   [N x L]
    gemm(stream, qph, kph, nullptr, nullptr, big, nullptr, nullptr,
         NNODES, LTOK, HDIM, 512, 512, LTOK, 0,
         sQb, sLb, (long)HEADS * sS2h, HEADS, HDIM, HDIM, sS2h,
         BATCH * HEADS, 0.125f, ACT_NONE);
    softmax_kernel<<<dim3(BATCH * HEADS * NNODES), dim3(128), 0, stream>>>(big, LTOK, Ph);

    _Float16* o2h = slotA;  // o consumed
    gemm(stream, Ph, vpT, nullptr, nullptr, nullptr, o2h, nullptr,
         NNODES, HDIM, LTOK, LTOK, LTOK, 512, 0,
         (long)HEADS * sS2h, (long)D_MODEL * LTOK, sQb, HEADS,
         sS2h, (long)HDIM * LTOK, HDIM,
         BATCH * HEADS, 1.0f, ACT_NONE);
    _Float16* geh = slotB;  // gc consumed
    gemm(stream, o2h, cmOutT, cmOutb, nullptr, nullptr, geh, nullptr,
         MN, 512, 512, 512, 512, 512, 0, 0,0,0, 1, 0,0,0, 1, 1.0f, ACT_NONE);
    gemm(stream, geh, goutT, goutb, nullptr, bufP, nullptr, nullptr,
         MN, 512, 512, 512, 512, 512, 0, 0,0,0, 1, 0,0,0, 1, 1.0f, ACT_NONE);
    ln_res_kernel<<<dim3(MN), dim3(128), 0, stream>>>(h1f, bufP, gng, gnb, h2f, hSlot);

    // ---- 3. gate (GRU path & attended memory are dead; attended == 0) ----
    _Float16* uxh = slotC;  // qp consumed
    gemm(stream, hSlot, gateT, gateb, h2f, uxf, uxh, nullptr,
         MN, 512, 512, 512, 512, 512, 0, 0,0,0, 1, 0,0,0, 1, 1.0f, ACT_GATE);

    // ---- 4. FFN + final LN ----
    _Float16* th = Ph;      // P mirrors consumed
    gemm(stream, uxh, ff1T, ff1b, nullptr, nullptr, th, nullptr,
         MN, FFD, 512, 512, 512, FFD, 0, 0,0,0, 1, 0,0,0, 1, 1.0f, ACT_GELU);
    gemm(stream, th, ff2T, ff2b, nullptr, bufP, nullptr, nullptr,
         MN, 512, FFD, FFD, FFD, 512, 0, 0,0,0, 1, 0,0,0, 1, 1.0f, ACT_NONE);
    ln_res_kernel<<<dim3(MN), dim3(128), 0, stream>>>(uxf, bufP, ln2g, ln2b, out, nullptr);
}
